// FinalizedQuantizedLinear_31387620999494
// MI455X (gfx1250) — matrix operand order repaired and checked by reading, not compiled, not run on hardware
//
#include <hip/hip_runtime.h>

typedef __attribute__((ext_vector_type(16))) __bf16        v16bf;
typedef __attribute__((ext_vector_type(8)))  float          v8f;
typedef __attribute__((ext_vector_type(4)))  float          v4f;
typedef __attribute__((ext_vector_type(4)))  unsigned short v4u;
typedef __attribute__((ext_vector_type(8)))  unsigned short v8u;
typedef __attribute__((ext_vector_type(16))) unsigned short v16u;

#define M_DIM 8192
#define K_DIM 4096
#define N_DIM 11008
#define NUM_CB 2
#define CB_SIZE 256
#define IN_G 8
#define NUM_IN_GROUPS (K_DIM / IN_G)   // 512

#define BM 128
#define BN 128
#define BK 64
#define LDSS (BK + 8)                  // 72 shorts = 144B row stride, conflict-free

__device__ __forceinline__ unsigned short f32_to_bf16(float f) {
  unsigned u = __builtin_bit_cast(unsigned, f);
  u += 0x7FFFu + ((u >> 16) & 1u);     // round-to-nearest-even
  return (unsigned short)(u >> 16);
}

__global__ __launch_bounds__(256)
void aqlm_wmma_gemm(const float* __restrict__ x,
                    const int*   __restrict__ codes,
                    const float* __restrict__ codebooks,
                    const float* __restrict__ scales,
                    const float* __restrict__ bias,
                    float* __restrict__ out)
{
  __shared__ __align__(16) unsigned short As[BM][LDSS];   // x tile, bf16
  __shared__ __align__(16) unsigned short Bs[BN][LDSS];   // dequantized W tile, bf16
  __shared__ __align__(16) float Cb[NUM_CB * CB_SIZE * IN_G]; // 16 KB codebooks

  const int t    = threadIdx.x;
  const int lane = t & 31;
  const int wave = t >> 5;
  const int wm   = (wave & 3) * 32;   // wave M offset within tile
  const int wn   = (wave >> 2) * 64;  // wave N offset within tile

  const int ntile = blockIdx.x * BN;
  const int mtile = blockIdx.y * BM;

  // ---- stage codebooks to LDS (4096 floats, 16B vectors) ----
  {
    const v4f* src = (const v4f*)codebooks;
    v4f*       dst = (v4f*)Cb;
    #pragma unroll
    for (int i = 0; i < (NUM_CB * CB_SIZE * IN_G / 4) / 256; ++i)
      dst[t + i * 256] = src[t + i * 256];
  }

  v8f acc[2][4];
  {
    v8f z = {};
    #pragma unroll
    for (int i = 0; i < 2; ++i)
      #pragma unroll
      for (int j = 0; j < 4; ++j) acc[i][j] = z;
  }

  for (int k0 = 0; k0 < K_DIM; k0 += BK) {
    __syncthreads();   // covers codebook staging (1st iter) + LDS reuse

    // ---- A tile: 128x64 f32 -> bf16 LDS (8 x float4 per thread) ----
    #pragma unroll
    for (int i = 0; i < 8; ++i) {
      int idx = t + i * 256;             // 0..2047
      int row = idx >> 4;                // 16 float4 per row
      int col = (idx & 15) << 2;
      v4f v = *(const v4f*)(x + (size_t)(mtile + row) * K_DIM + k0 + col);
      v4u h;
      h.x = f32_to_bf16(v.x); h.y = f32_to_bf16(v.y);
      h.z = f32_to_bf16(v.z); h.w = f32_to_bf16(v.w);
      *(v4u*)&As[row][col] = h;
    }

    // ---- B tile: dequantize 128 rows x 8 in-groups -> bf16 LDS ----
    #pragma unroll
    for (int i = 0; i < 4; ++i) {
      int idx = t + i * 256;             // 0..1023
      int o   = idx >> 3;                // 0..127
      int g   = idx & 7;                 // in-group within K-tile
      int oo  = ntile + o;
      const int* cp = codes + ((size_t)oo * NUM_IN_GROUPS + (k0 >> 3) + g) * NUM_CB;
      int   c0 = cp[0];
      int   c1 = cp[1];
      float s  = scales[oo];
      const float* w0 = Cb + (c0 << 3);
      const float* w1 = Cb + (CB_SIZE * IN_G) + (c1 << 3);
      v8u packed;
      #pragma unroll
      for (int j = 0; j < 8; ++j)
        packed[j] = f32_to_bf16((w0[j] + w1[j]) * s);
      *(v8u*)&Bs[o][g << 3] = packed;    // 16B store, K-contiguous row of W
    }
    __syncthreads();

    // ---- compute: two K=32 steps, 16 WMMAs per wave per K-tile ----
    #pragma unroll
    for (int ks = 0; ks < BK; ks += 32) {
      v16bf afrag[2];
      #pragma unroll
      for (int tm = 0; tm < 2; ++tm) {
        int row = wm + tm * 16 + (lane & 15);
        int kc  = ks + ((lane >> 4) << 3);          // +0 or +8
        v8u lo = *(const v8u*)&As[row][kc];         // K chunk [kc..kc+7]
        v8u hi = *(const v8u*)&As[row][kc + 16];    // K chunk [kc+16..kc+23]
        v16u cat = __builtin_shufflevector(lo, hi,
            0,1,2,3,4,5,6,7,8,9,10,11,12,13,14,15);
        afrag[tm] = __builtin_bit_cast(v16bf, cat);
      }
      v16bf bfrag[4];
      #pragma unroll
      for (int tn = 0; tn < 4; ++tn) {
        int col = wn + tn * 16 + (lane & 15);
        int kc  = ks + ((lane >> 4) << 3);          // +0 or +8
        v8u lo = *(const v8u*)&Bs[col][kc];         // K chunk [kc..kc+7]
        v8u hi = *(const v8u*)&Bs[col][kc + 16];    // K chunk [kc+16..kc+23]
        v16u cat = __builtin_shufflevector(lo, hi,
            0,1,2,3,4,5,6,7,8,9,10,11,12,13,14,15);
        bfrag[tn] = __builtin_bit_cast(v16bf, cat);
      }
      #pragma unroll
      for (int tm = 0; tm < 2; ++tm)
        #pragma unroll
        for (int tn = 0; tn < 4; ++tn)
          acc[tm][tn] = __builtin_amdgcn_wmma_f32_16x16x32_bf16(
              false, afrag[tm], false, bfrag[tn],
              (short)0, acc[tm][tn], false, false);
    }
  }

  // ---- epilogue: C/D layout (VGPR r: lanes0-15 -> M=r, lanes16-31 -> M=r+8) ----
  #pragma unroll
  for (int tn = 0; tn < 4; ++tn) {
    int n = ntile + wn + tn * 16 + (lane & 15);
    float bv = bias[n];
    #pragma unroll
    for (int tm = 0; tm < 2; ++tm) {
      int mbase = mtile + wm + tm * 16 + ((lane >> 4) << 3);
      #pragma unroll
      for (int r = 0; r < 8; ++r)
        out[(size_t)(mbase + r) * N_DIM + n] = acc[tm][tn][r] + bv;
    }
  }
}

extern "C" void kernel_launch(void* const* d_in, const int* in_sizes, int n_in,
                              void* d_out, int out_size, void* d_ws, size_t ws_size,
                              hipStream_t stream) {
  (void)in_sizes; (void)n_in; (void)out_size; (void)d_ws; (void)ws_size;
  const float* x         = (const float*)d_in[0];
  const int*   codes     = (const int*)  d_in[1];
  const float* codebooks = (const float*)d_in[2];
  const float* scales    = (const float*)d_in[3];
  const float* bias      = (const float*)d_in[4];
  float*       out       = (float*)d_out;

  dim3 grid(N_DIM / BN, M_DIM / BM);   // 86 x 64 blocks
  aqlm_wmma_gemm<<<grid, 256, 0, stream>>>(x, codes, codebooks, scales, bias, out);
}
